// BitCNNFrontend_77764677861390
// MI455X (gfx1250) — compile-verified, matrix operand order checked by reading, not run on hardware
//
#include <hip/hip_runtime.h>
#include <hip/hip_bf16.h>

typedef __attribute__((ext_vector_type(16))) _Float16 v16h;
typedef __attribute__((ext_vector_type(8)))  _Float16 h8;
typedef __attribute__((ext_vector_type(4)))  _Float16 h4;
typedef __attribute__((ext_vector_type(8)))  float    v8f;
typedef __attribute__((ext_vector_type(4)))  unsigned int u32x4;
typedef __attribute__((ext_vector_type(4)))  int      i32x4;
typedef __attribute__((ext_vector_type(8)))  int      i32x8;

// ---------------------------------------------------------------------------
// Weight ternary quantization: scale = mean(|W|) + 1e-8 ; Wq = clip(rint(W/s),-1,1)*s
// ---------------------------------------------------------------------------
__global__ void absmean_kernel(const float* __restrict__ w, float* __restrict__ scale, int nel) {
    __shared__ float sh[256];
    float s = 0.f;
    for (int i = threadIdx.x; i < nel; i += 256) s += fabsf(w[i]);
    sh[threadIdx.x] = s;
    __syncthreads();
    for (int o = 128; o > 0; o >>= 1) {
        if (threadIdx.x < o) sh[threadIdx.x] += sh[threadIdx.x + o];
        __syncthreads();
    }
    if (threadIdx.x == 0) scale[0] = sh[0] / (float)nel + 1e-8f;
}

__global__ void quantize_kernel(const float* __restrict__ w, const float* __restrict__ scale,
                                _Float16* __restrict__ wq, int nel) {
    int i = blockIdx.x * blockDim.x + threadIdx.x;
    if (i >= nel) return;
    float sc = scale[0];
    float q = rintf(w[i] / sc);
    q = fminf(1.f, fmaxf(-1.f, q));
    wq[i] = (_Float16)(q * sc);
}

// ---------------------------------------------------------------------------
// Layer 1: cin=1, cout=32, k=10, s=5, pad=5  (K too thin for WMMA -> direct)
// ---------------------------------------------------------------------------
__global__ void conv1_direct(const float* __restrict__ X, const _Float16* __restrict__ Wq,
                             float* __restrict__ Y, int Lin, int Lout) {
    int idx = blockIdx.x * blockDim.x + threadIdx.x;     // over cout*Lout
    int n = blockIdx.y;
    if (idx >= 32 * Lout) return;
    int co = idx / Lout;
    int l  = idx - co * Lout;
    const float* xn = X + (size_t)n * Lin;
    int base = l * 5 - 5;
    float s = 0.f;
#pragma unroll
    for (int kk = 0; kk < 10; ++kk) {
        int p = base + kk;
        float wv = (float)Wq[co * 10 + kk];
        if (p >= 0 && p < Lin) s += wv * xn[p];
    }
    Y[((size_t)n * 32 + co) * Lout + l] = s;
}

// ---------------------------------------------------------------------------
// Layers 2-4: implicit GEMM via v_wmma_f32_16x16x32_f16.
//   M = cout tile of 16, N = 16 output positions, K = cin*8 in chunks of 32.
//   Weight tile (16 x KTOT f16) is DMA'd into LDS once per block by the
//   Tensor Data Mover; A fragments then come from LDS as 16B ds loads,
//   B fragments come from global as aligned 8B loads (contiguous in X).
// ---------------------------------------------------------------------------
template<int CIN, int COUT>
__global__ __launch_bounds__(128) void conv_wmma_k8s4(
    const _Float16* __restrict__ Xh, const _Float16* __restrict__ Wq,
    float* __restrict__ Y, int Lin, int Lout)
{
    constexpr int KTOT = CIN * 8;
    __shared__ __align__(16) _Float16 aTile[16 * KTOT];

    const int wave = threadIdx.x >> 5;
    const int lane = threadIdx.x & 31;
    const int ltile = blockIdx.x * 4 + wave;          // wave-uniform
    const int co0 = blockIdx.y * 16;
    const int n   = blockIdx.z;

    // ---- TDM: DMA the 16 x KTOT weight tile (row-major) into LDS ----
    if (wave == 0) {
        const unsigned long long gaddr =
            (unsigned long long)(uintptr_t)(Wq + (size_t)co0 * KTOT);
        const unsigned int ldsa = (unsigned int)(uintptr_t)(&aTile[0]);
        // D# group0: count=1 | lds_addr | global_addr[56:0] | type=2
        u32x4 g0 = { 1u,
                     ldsa,
                     (unsigned int)(gaddr & 0xffffffffu),
                     (unsigned int)((gaddr >> 32) & 0x01ffffffu) | (2u << 30) };
        // D# group1: data_size=2B; tensor_dim0=KTOT; tensor_dim1=COUT;
        //            tile_dim0=KTOT; tile_dim1=16; tensor_dim0_stride=KTOT
        i32x8 g1 = { (int)(1u << 16),
                     (int)(((unsigned)KTOT & 0xffffu) << 16),
                     (int)(((unsigned)COUT & 0xffffu) << 16),
                     (int)((unsigned)KTOT << 16),
                     16,
                     KTOT,
                     0, 0 };
        i32x4 z4 = { 0, 0, 0, 0 };
#if defined(__clang_major__) && (__clang_major__ >= 23)
        i32x8 z8 = { 0, 0, 0, 0, 0, 0, 0, 0 };
        __builtin_amdgcn_tensor_load_to_lds(g0, g1, z4, z4, z8, 0);
#else
        __builtin_amdgcn_tensor_load_to_lds(g0, g1, z4, z4, 0);
#endif
        __builtin_amdgcn_s_wait_tensorcnt(0);
    }
    __syncthreads();

    if (ltile * 16 >= Lout) return;   // wave-uniform; wave 0 never exits early

    const int m     = lane & 15;      // A row / B col / D col
    const int khalf = lane >> 4;      // lane group selects K sub-range

    const _Float16* Xn = Xh + (size_t)n * CIN * Lin;
    const int l = ltile * 16 + m;
    const int basepos = l * 4 - 4;                       // 8-byte aligned in f16
    const bool safe = (basepos >= 0) && (basepos + 7 < Lin);

    const _Float16* arow = aTile + m * KTOT + khalf * 8; // 16B aligned (+q0/+16)

    v8f acc = {};
    for (int q0 = 0; q0 < KTOT; q0 += 32) {
        const int ciA = (q0 >> 3) + khalf;   // K-chunk rows for this lane half
        const int ciB = ciA + 2;
        const _Float16* xa = Xn + (size_t)ciA * Lin;
        const _Float16* xb = Xn + (size_t)ciB * Lin;
        __builtin_prefetch(xa + (size_t)4 * Lin, 0, 1);  // next chunk's rows

        // A fragment: two aligned 16B LDS reads
        h8 alo = *(const h8*)(arow + q0);
        h8 ahi = *(const h8*)(arow + q0 + 16);
        v16h a = __builtin_shufflevector(alo, ahi,
                 0, 1, 2, 3, 4, 5, 6, 7, 8, 9, 10, 11, 12, 13, 14, 15);

        // B fragment: four aligned 8B global reads (contiguous kk window)
        v16h b;
        if (safe) {
            h4 x0 = *(const h4*)(xa + basepos);
            h4 x1 = *(const h4*)(xa + basepos + 4);
            h4 x2 = *(const h4*)(xb + basepos);
            h4 x3 = *(const h4*)(xb + basepos + 4);
            h8 blo = __builtin_shufflevector(x0, x1, 0, 1, 2, 3, 4, 5, 6, 7);
            h8 bhi = __builtin_shufflevector(x2, x3, 0, 1, 2, 3, 4, 5, 6, 7);
            b = __builtin_shufflevector(blo, bhi,
                0, 1, 2, 3, 4, 5, 6, 7, 8, 9, 10, 11, 12, 13, 14, 15);
        } else {
#pragma unroll
            for (int kk = 0; kk < 8; ++kk) {
                int p = basepos + kk;
                bool in = (p >= 0) && (p < Lin);
                b[kk]     = in ? xa[p] : (_Float16)0.f;
                b[kk + 8] = in ? xb[p] : (_Float16)0.f;
            }
        }

        acc = __builtin_amdgcn_wmma_f32_16x16x32_f16(
            /*neg_a=*/false, a, /*neg_b=*/false, b,
            /*c_mod=*/(short)0, acc, /*reuse_a=*/false, /*reuse_b=*/false);
    }

    float* Yn = Y + (size_t)n * COUT * Lout;
    const int lc = ltile * 16 + m;
#pragma unroll
    for (int r = 0; r < 8; ++r) {
        const int co = co0 + r + khalf * 8;
        if (lc < Lout) Yn[(size_t)co * Lout + lc] = acc[r];
    }
}

// ---------------------------------------------------------------------------
// Per-sample layernorm stats over (C,L): deterministic two-stage reduction.
// ---------------------------------------------------------------------------
__global__ void reduce_stats_partial(const float* __restrict__ Y, float* __restrict__ part,
                                     long long total, int nblocks) {
    const int n = blockIdx.y;
    const float* y = Y + (size_t)n * total;
    long long i = (long long)blockIdx.x * blockDim.x + threadIdx.x;
    const long long stride = (long long)nblocks * blockDim.x;
    float s = 0.f, ss = 0.f;
    for (; i < total; i += stride) { float v = y[i]; s += v; ss += v * v; }
    __shared__ float sh[256], sh2[256];
    sh[threadIdx.x] = s; sh2[threadIdx.x] = ss;
    __syncthreads();
    for (int o = 128; o > 0; o >>= 1) {
        if (threadIdx.x < o) { sh[threadIdx.x] += sh[threadIdx.x + o]; sh2[threadIdx.x] += sh2[threadIdx.x + o]; }
        __syncthreads();
    }
    if (threadIdx.x == 0) {
        part[(size_t)(n * nblocks + blockIdx.x) * 2 + 0] = sh[0];
        part[(size_t)(n * nblocks + blockIdx.x) * 2 + 1] = sh2[0];
    }
}

__global__ void finalize_stats(const float* __restrict__ part, float* __restrict__ stats,
                               int nblocks, float inv_count) {
    const int n = blockIdx.x;
    float s = 0.f, ss = 0.f;
    for (int i = threadIdx.x; i < nblocks; i += blockDim.x) {
        s  += part[(size_t)(n * nblocks + i) * 2 + 0];
        ss += part[(size_t)(n * nblocks + i) * 2 + 1];
    }
    __shared__ float sh[256], sh2[256];
    sh[threadIdx.x] = s; sh2[threadIdx.x] = ss;
    __syncthreads();
    for (int o = 128; o > 0; o >>= 1) {
        if (threadIdx.x < o) { sh[threadIdx.x] += sh[threadIdx.x + o]; sh2[threadIdx.x] += sh2[threadIdx.x + o]; }
        __syncthreads();
    }
    if (threadIdx.x == 0) {
        float mu  = sh[0] * inv_count;
        float var = sh2[0] * inv_count - mu * mu;
        stats[n * 2 + 0] = mu;
        stats[n * 2 + 1] = rsqrtf(var + 1e-5f);
    }
}

// ---------------------------------------------------------------------------
// normalize + affine + snake; emit f16 next-layer input, or transposed f32 out.
// ---------------------------------------------------------------------------
template<bool LAST>
__global__ void norm_snake(const float* __restrict__ Y, const float* __restrict__ stats,
                           const float* __restrict__ g, const float* __restrict__ b,
                           const float* __restrict__ al, const float* __restrict__ ph,
                           _Float16* __restrict__ Hout, float* __restrict__ Dout,
                           int C, int Lout) {
    const int n = blockIdx.y;
    const long long total = (long long)C * Lout;
    const long long idx = (long long)blockIdx.x * blockDim.x + threadIdx.x;
    if (idx >= total) return;
    const int c = (int)(idx / Lout);
    const int l = (int)(idx - (long long)c * Lout);
    const float mu = stats[n * 2 + 0];
    const float rstd = stats[n * 2 + 1];
    const float v  = Y[(size_t)n * total + idx];
    const float yn = (v - mu) * rstd * g[c] + b[c];
    const float a  = al[c];
    const float s  = sinf(a * yn + ph[c]);
    const float out = yn + s * s / a;
    if (LAST) {
        Dout[((size_t)n * Lout + l) * C + c] = out;   // (N, L, C)
    } else {
        Hout[(size_t)n * total + idx] = (_Float16)out;
    }
}

// ---------------------------------------------------------------------------
extern "C" void kernel_launch(void* const* d_in, const int* in_sizes, int n_in,
                              void* d_out, int out_size, void* d_ws, size_t ws_size,
                              hipStream_t stream) {
    (void)in_sizes; (void)n_in; (void)out_size; (void)ws_size;

    const float* x = (const float*)d_in[0];
    const float* W[4]  = {(const float*)d_in[1],  (const float*)d_in[6],
                          (const float*)d_in[11], (const float*)d_in[16]};
    const float* G[4]  = {(const float*)d_in[2],  (const float*)d_in[7],
                          (const float*)d_in[12], (const float*)d_in[17]};
    const float* Bt[4] = {(const float*)d_in[3],  (const float*)d_in[8],
                          (const float*)d_in[13], (const float*)d_in[18]};
    const float* Al[4] = {(const float*)d_in[4],  (const float*)d_in[9],
                          (const float*)d_in[14], (const float*)d_in[19]};
    const float* Ph[4] = {(const float*)d_in[5],  (const float*)d_in[10],
                          (const float*)d_in[15], (const float*)d_in[20]};

    const int N = 32;
    const int L0 = 320000, L1 = 64001, L2 = 16001, L3 = 4001, L4 = 1001;
    const int WNEL[4] = {32 * 1 * 10, 64 * 32 * 8, 128 * 64 * 8, 256 * 128 * 8};
    const int RB = 64;  // reduction blocks per sample

    // ---- workspace carve (yslab reused each layer; h buffers ping-pong) ----
    char* base = (char*)d_ws;
    size_t off = 0;
    auto carve = [&](size_t bytes) -> char* {
        char* p = base + off;
        off += (bytes + 255) & ~(size_t)255;
        return p;
    };
    float*    yslab = (float*)   carve((size_t)N * 32 * L1 * sizeof(float));     // max y
    _Float16* hA    = (_Float16*)carve((size_t)N * 32 * L1 * sizeof(_Float16));  // h1 / h3
    _Float16* hB    = (_Float16*)carve((size_t)N * 64 * L2 * sizeof(_Float16));  // h2
    _Float16* wq[4];
    for (int i = 0; i < 4; ++i) wq[i] = (_Float16*)carve((size_t)WNEL[i] * sizeof(_Float16));
    float* scl   = (float*)carve(4 * sizeof(float));
    float* part  = (float*)carve((size_t)N * RB * 2 * sizeof(float));
    float* stats = (float*)carve((size_t)N * 2 * sizeof(float));

    // ---- quantize all weights ----
    for (int i = 0; i < 4; ++i) {
        absmean_kernel<<<1, 256, 0, stream>>>(W[i], scl + i, WNEL[i]);
        quantize_kernel<<<(WNEL[i] + 255) / 256, 256, 0, stream>>>(W[i], scl + i, wq[i], WNEL[i]);
    }

    // ---- layer 1 (direct) ----
    conv1_direct<<<dim3((32 * L1 + 255) / 256, N), 256, 0, stream>>>(x, wq[0], yslab, L0, L1);
    reduce_stats_partial<<<dim3(RB, N), 256, 0, stream>>>(yslab, part, (long long)32 * L1, RB);
    finalize_stats<<<N, 256, 0, stream>>>(part, stats, RB, 1.0f / (float)((long long)32 * L1));
    {
        long long tot = (long long)32 * L1;
        norm_snake<false><<<dim3((unsigned)((tot + 255) / 256), N), 256, 0, stream>>>(
            yslab, stats, G[0], Bt[0], Al[0], Ph[0], hA, nullptr, 32, L1);
    }

    // ---- layer 2 (WMMA): hA -> yslab -> hB ----
    {
        int tiles = (L2 + 15) / 16;
        conv_wmma_k8s4<32, 64><<<dim3((tiles + 3) / 4, 64 / 16, N), 128, 0, stream>>>(
            hA, wq[1], yslab, L1, L2);
        reduce_stats_partial<<<dim3(RB, N), 256, 0, stream>>>(yslab, part, (long long)64 * L2, RB);
        finalize_stats<<<N, 256, 0, stream>>>(part, stats, RB, 1.0f / (float)((long long)64 * L2));
        long long tot = (long long)64 * L2;
        norm_snake<false><<<dim3((unsigned)((tot + 255) / 256), N), 256, 0, stream>>>(
            yslab, stats, G[1], Bt[1], Al[1], Ph[1], hB, nullptr, 64, L2);
    }

    // ---- layer 3 (WMMA): hB -> yslab -> hA ----
    {
        int tiles = (L3 + 15) / 16;
        conv_wmma_k8s4<64, 128><<<dim3((tiles + 3) / 4, 128 / 16, N), 128, 0, stream>>>(
            hB, wq[2], yslab, L2, L3);
        reduce_stats_partial<<<dim3(RB, N), 256, 0, stream>>>(yslab, part, (long long)128 * L3, RB);
        finalize_stats<<<N, 256, 0, stream>>>(part, stats, RB, 1.0f / (float)((long long)128 * L3));
        long long tot = (long long)128 * L3;
        norm_snake<false><<<dim3((unsigned)((tot + 255) / 256), N), 256, 0, stream>>>(
            yslab, stats, G[2], Bt[2], Al[2], Ph[2], hA, nullptr, 128, L3);
    }

    // ---- layer 4 (WMMA): hA -> yslab -> d_out (transposed f32) ----
    {
        int tiles = (L4 + 15) / 16;
        conv_wmma_k8s4<128, 256><<<dim3((tiles + 3) / 4, 256 / 16, N), 128, 0, stream>>>(
            hA, wq[3], yslab, L3, L4);
        reduce_stats_partial<<<dim3(RB, N), 256, 0, stream>>>(yslab, part, (long long)256 * L4, RB);
        finalize_stats<<<N, 256, 0, stream>>>(part, stats, RB, 1.0f / (float)((long long)256 * L4));
        long long tot = (long long)256 * L4;
        norm_snake<true><<<dim3((unsigned)((tot + 255) / 256), N), 256, 0, stream>>>(
            yslab, stats, G[3], Bt[3], Al[3], Ph[3], nullptr, (float*)d_out, 256, L4);
    }
}
